// CRF_74526272520633
// MI455X (gfx1250) — matrix operand order repaired and checked by reading, not compiled, not run on hardware
//
#include <hip/hip_runtime.h>

#define Bsz 64
#define Ssz 512
#define Tsz 64

#define LOG2E 1.4426950408889634f
#define LN2   0.6931471805599453f

typedef _Float16 v16h __attribute__((ext_vector_type(16)));
typedef __fp16   fp16x2 __attribute__((ext_vector_type(2)));
typedef float    v8f  __attribute__((ext_vector_type(8)));
typedef float    f4   __attribute__((ext_vector_type(4)));

union H16 { _Float16 h[16]; v16h v; };
union U8V { unsigned  u[8];  v16h v; };
union HU  { fp16x2 h; unsigned u; };

__device__ __forceinline__ float shflxf(float x, int m) { return __shfl_xor(x, m, 32); }

// pack two floats to f16 pair (v_cvt_pkrtz_f16_f32)
__device__ __forceinline__ unsigned pk2(float x0, float x1) {
  HU u;
  u.h = __builtin_amdgcn_cvt_pkrtz(x0, x1);
  return u.u;
}

// ---------------------------------------------------------------------------
// Forward (partition) scan: 4 blocks x 1 wave; wave owns 16 batches, all tags.
// logsumexp as centered-exp GEMM on v_wmma_f32_16x16x32_f16.
//
// Tag permutation: register slot (mt, r, lane-half hi) holds tag
//   tau = 32*mt[1] + 16*hi + 8*mt[0] + r
// so the next step's B operand is a static repack of the lane's own registers.
//
// Exp-domain state: instead of part we keep E = exp(part - c) packed in the
// f16 B operands themselves.  Per step: E_new = D * exp(cm + f - Delta) with
// Delta = c_s - c_{s-1} = ln(max Eu_{s-1}) computed one step late (scalar
// v_log per lane, off the critical path).  cAcc accumulates the center;
// frozen (masked) lanes keep E and cAcc bit-exact via dword selects and a
// mask-gated fma.  part is reconstructed only at finalize.
// ---------------------------------------------------------------------------
__global__ __launch_bounds__(32, 1)
void crf_forward_kernel(const float* __restrict__ feats,
                        const float* __restrict__ trans,
                        const int*   __restrict__ mask,
                        float*       __restrict__ fpart) {
  const int lane = threadIdx.x;          // 0..31
  const int lm   = lane & 15;            // batch within wave, also N index
  const int hi   = (lane >> 4) & 1;      // lane half
  const int b    = blockIdx.x * 16 + lm; // global batch this lane serves

  // ---- global centering constant C = max(trans)
  float cm = -3.0e38f;
  for (int i = lane; i < Tsz * Tsz; i += 32) cm = fmaxf(cm, trans[i]);
  #pragma unroll
  for (int d = 1; d < 32; d <<= 1) cm = fmaxf(cm, shflxf(cm, d));

  // ---- constant A fragments with permuted M rows:
  // A[m][k] = exp(trans[k][tau(m)] - C), m = 16*mt + (lane%16),
  // element e -> K offset = e + 16*half   (contiguous K order, matches B repack)
  v16h A[4][2];
  #pragma unroll
  for (int mt = 0; mt < 4; ++mt) {
    const int tagOut = 32 * (mt >> 1) + 16 * ((lm >> 3) & 1) + 8 * (mt & 1) + (lm & 7);
    #pragma unroll
    for (int kc = 0; kc < 2; ++kc) {
      H16 af;
      #pragma unroll
      for (int e = 0; e < 16; ++e) {
        int koff = e + 16 * hi;
        int k = 32 * kc + koff;
        af.h[e] = (_Float16)__expf(trans[k * Tsz + tagOut] - cm);
      }
      A[mt][kc] = af.v;
    }
  }

  // per-tile base tag for this lane half: tags tb..tb+7 are 8 consecutive
  #define TBASE(mt) (32 * ((mt) >> 1) + 8 * ((mt) & 1) + 16 * hi)
  // packed-state dword index for (tile mt, pair v): B0 = tiles 0,1; B1 = 2,3
  #define BIDX(mt, v) (((mt) >> 1) * 8 + ((mt) & 1) * 4 + (v))

  // ---- init: part0 = feats[b,0,:] + trans[START,:]; E = exp(part0 - c0)
  U8V B0u, B1u;        // persistent exp-domain state (= WMMA B operands)
  float cAcc;          // accumulated center (per lane, both halves equal)
  float Delta = 0.0f;  // center increment to apply this step (lags one step)
  {
    float p0[4][8];
    const float* frow0 = feats + (size_t)b * Ssz * Tsz;
    const float* trow  = trans + (Tsz - 2) * Tsz;   // START row
    #pragma unroll
    for (int mt = 0; mt < 4; ++mt) {
      int tb = TBASE(mt);
      f4 f0 = *(const f4*)(frow0 + tb);
      f4 f1 = *(const f4*)(frow0 + tb + 4);
      f4 t0 = *(const f4*)(trow + tb);
      f4 t1 = *(const f4*)(trow + tb + 4);
      #pragma unroll
      for (int j = 0; j < 4; ++j) {
        p0[mt][j]     = f0[j] + t0[j];
        p0[mt][4 + j] = f1[j] + t1[j];
      }
    }
    float m = p0[0][0];
    #pragma unroll
    for (int mt = 0; mt < 4; ++mt)
      #pragma unroll
      for (int r = 0; r < 8; ++r) m = fmaxf(m, p0[mt][r]);
    m = fmaxf(m, shflxf(m, 16));
    cAcc = m;
    float negc2 = -m * LOG2E;
    #pragma unroll
    for (int mt = 0; mt < 4; ++mt)
      #pragma unroll
      for (int v = 0; v < 4; ++v) {
        unsigned pn = pk2(__builtin_amdgcn_exp2f(__builtin_fmaf(p0[mt][2*v],   LOG2E, negc2)),
                          __builtin_amdgcn_exp2f(__builtin_fmaf(p0[mt][2*v+1], LOG2E, negc2)));
        if (mt < 2) B0u.u[BIDX(mt, v)] = pn; else B1u.u[BIDX(mt, v) - 8] = pn;
      }
  }

  const float cmL2E = cm * LOG2E;

  // warm prefetch window (global_prefetch_b8: no counters, no VGPR return)
  const int PF = 8;
  #pragma unroll
  for (int s = 1; s <= PF; ++s) {
    const char* p = (const char*)(feats + ((size_t)b * Ssz + s) * Tsz) + hi * 128;
    __builtin_prefetch(p, 0, 0);
  }

  // running pointers (advance by one step each iteration)
  const float* frow = feats + ((size_t)b * Ssz + 1) * Tsz;
  const char*  prow = (const char*)(feats + ((size_t)b * Ssz + 1 + PF) * Tsz) + hi * 128;
  const int*   mrow = mask + b * Ssz + 1;

  #pragma unroll 1
  for (int s = 1; s < Ssz; ++s) {
    if (s + PF < Ssz) __builtin_prefetch(prow, 0, 0);
    int mk = *mrow;                          // 0/1, per batch (prefix mask)
    unsigned mbits = (unsigned)(-mk);
    float fmk = (float)mk;

    // ---- D[mt] = expT_perm^T x E  (f32 accumulate over 2 K-chunks)
    v8f d[4];
    #pragma unroll
    for (int mt = 0; mt < 4; ++mt) {
      v8f c;
      #pragma unroll
      for (int i = 0; i < 8; ++i) c[i] = 0.0f;
      c = __builtin_amdgcn_wmma_f32_16x16x32_f16(false, A[mt][0], false, B0u.v,
                                                 (short)0, c, false, false);
      c = __builtin_amdgcn_wmma_f32_16x16x32_f16(false, A[mt][1], false, B1u.v,
                                                 (short)0, c, false, false);
      d[mt] = c;
    }

    // ---- emissions for this step (prefetched; b128 loads)
    float fb[4][8];
    #pragma unroll
    for (int mt = 0; mt < 4; ++mt) {
      int tb = TBASE(mt);
      f4 f0 = *(const f4*)(frow + tb);
      f4 f1 = *(const f4*)(frow + tb + 4);
      #pragma unroll
      for (int j = 0; j < 4; ++j) { fb[mt][j] = f0[j]; fb[mt][4 + j] = f1[j]; }
    }

    // ---- Eu = D * exp(cm + f - Delta)   (no log on the critical path)
    float scal2 = __builtin_fmaf(-Delta, LOG2E, cmL2E);
    float Eu[4][8];
    #pragma unroll
    for (int mt = 0; mt < 4; ++mt)
      #pragma unroll
      for (int r = 0; r < 8; ++r)
        Eu[mt][r] = d[mt][r] *
            __builtin_amdgcn_exp2f(__builtin_fmaf(fb[mt][r], LOG2E, scal2));

    // ---- pack + masked select on packed dwords (frozen lanes keep state)
    #pragma unroll
    for (int mt = 0; mt < 4; ++mt)
      #pragma unroll
      for (int v = 0; v < 4; ++v) {
        unsigned pn = pk2(Eu[mt][2*v], Eu[mt][2*v+1]);
        if (mt < 2) {
          unsigned o = B0u.u[BIDX(mt, v)];
          B0u.u[BIDX(mt, v)] = (pn & mbits) | (o & ~mbits);
        } else {
          unsigned o = B1u.u[BIDX(mt, v) - 8];
          B1u.u[BIDX(mt, v) - 8] = (pn & mbits) | (o & ~mbits);
        }
      }

    // ---- center bookkeeping (scalar per lane; Delta lags one step)
    cAcc = __builtin_fmaf(fmk, Delta, cAcc);
    {
      float m = Eu[0][0];
      #pragma unroll
      for (int mt = 0; mt < 4; ++mt)
        #pragma unroll
        for (int r = 0; r < 8; ++r) m = fmaxf(m, Eu[mt][r]);
      m = fmaxf(m, shflxf(m, 16));
      Delta = fmk * (LN2 * __builtin_amdgcn_logf(m));   // 0 once frozen
    }

    frow += Tsz;
    prow += Tsz * sizeof(float);
    mrow += 1;
  }

  // ---- finalize: part = cAcc + ln2*log2(E);
  //      forward[b] = cAcc + lse_tags(ln(E) + trans[tag][END]); sum batches
  float val[4][8];
  #pragma unroll
  for (int mt = 0; mt < 4; ++mt) {
    int tb = TBASE(mt);
    #pragma unroll
    for (int r = 0; r < 8; ++r) {
      HU u;
      u.u = (mt < 2) ? B0u.u[BIDX(mt, r >> 1)] : B1u.u[BIDX(mt, r >> 1) - 8];
      float e = (float)u.h[r & 1];
      val[mt][r] = LN2 * __builtin_amdgcn_logf(e) + trans[(tb + r) * Tsz + (Tsz - 1)];
    }
  }
  float m = val[0][0];
  #pragma unroll
  for (int mt = 0; mt < 4; ++mt)
    #pragma unroll
    for (int r = 0; r < 8; ++r) m = fmaxf(m, val[mt][r]);
  float ss = 0.0f;
  #pragma unroll
  for (int mt = 0; mt < 4; ++mt)
    #pragma unroll
    for (int r = 0; r < 8; ++r) ss += __expf(val[mt][r] - m);
  float om = shflxf(m, 16), os = shflxf(ss, 16);
  float M2 = fmaxf(m, om);
  float S2 = ss * __expf(m - M2) + os * __expf(om - M2);
  float fsc = cAcc + M2 + __logf(S2);    // per-batch forward score
  #pragma unroll
  for (int d2 = 1; d2 <= 8; d2 <<= 1) fsc += shflxf(fsc, d2);  // sum 16 batches
  if (lane == 0) fpart[blockIdx.x] = fsc;
  #undef TBASE
  #undef BIDX
}

// ---------------------------------------------------------------------------
// Gold path score per batch (deterministic fixed-order block reduction).
// ---------------------------------------------------------------------------
__global__ __launch_bounds__(256)
void crf_gold_kernel(const float* __restrict__ feats,
                     const float* __restrict__ trans,
                     const int*   __restrict__ mask,
                     const int*   __restrict__ tags,
                     float*       __restrict__ gpart) {
  const int b = blockIdx.x;
  const int tid = threadIdx.x;
  __shared__ float sacc[256];
  __shared__ int   scnt[256];
  float acc = 0.0f;
  int cnt = 0;
  for (int s = tid; s < Ssz; s += 256) {
    int mv = mask[b * Ssz + s];
    if (mv) {
      int tg = tags[b * Ssz + s];
      int pv = (s == 0) ? (Tsz - 2) : tags[b * Ssz + s - 1];
      acc += feats[((size_t)b * Ssz + s) * Tsz + tg] + trans[pv * Tsz + tg];
      cnt += 1;
    }
  }
  sacc[tid] = acc;
  scnt[tid] = cnt;
  __syncthreads();
  for (int off = 128; off > 0; off >>= 1) {
    if (tid < off) { sacc[tid] += sacc[tid + off]; scnt[tid] += scnt[tid + off]; }
    __syncthreads();
  }
  if (tid == 0) {
    int len = scnt[0];
    int eid = tags[b * Ssz + len - 1];
    gpart[b] = sacc[0] + trans[eid * Tsz + (Tsz - 1)];
  }
}

// ---------------------------------------------------------------------------
// Fixed-order scalar combine: out = sum(forward partials) - sum(gold partials)
// ---------------------------------------------------------------------------
__global__ void crf_finalize_kernel(const float* __restrict__ fpart,
                                    const float* __restrict__ gpart,
                                    float* __restrict__ out) {
  if (threadIdx.x == 0 && blockIdx.x == 0) {
    float f = 0.0f;
    for (int i = 0; i < 4; ++i) f += fpart[i];
    float g = 0.0f;
    for (int i = 0; i < Bsz; ++i) g += gpart[i];
    out[0] = f - g;
  }
}

extern "C" void kernel_launch(void* const* d_in, const int* in_sizes, int n_in,
                              void* d_out, int out_size, void* d_ws, size_t ws_size,
                              hipStream_t stream) {
  const float* feats = (const float*)d_in[0];  // [B,S,T] f32
  const float* trans = (const float*)d_in[1];  // [T,T]   f32
  const int*   mask  = (const int*)d_in[2];    // [B,S]   i32
  const int*   tags  = (const int*)d_in[3];    // [B,S]   i32
  float* ws    = (float*)d_ws;
  float* fpart = ws;       // 4 floats: per-workgroup forward partials
  float* gpart = ws + 4;   // 64 floats: per-batch gold partials

  crf_forward_kernel<<<4, 32, 0, stream>>>(feats, trans, mask, fpart);
  crf_gold_kernel<<<Bsz, 256, 0, stream>>>(feats, trans, mask, tags, gpart);
  crf_finalize_kernel<<<1, 32, 0, stream>>>(fpart, gpart, (float*)d_out);
}
